// AnyAttention_2980707303965
// MI455X (gfx1250) — matrix operand order repaired and checked by reading, not compiled, not run on hardware
//
#include <hip/hip_runtime.h>
#include <hip/hip_bf16.h>

// ---- types for WMMA ----
typedef __attribute__((ext_vector_type(16))) _Float16 v16h;
typedef __attribute__((ext_vector_type(8)))  _Float16 h8;
typedef __attribute__((ext_vector_type(4)))  _Float16 h4;
typedef __attribute__((ext_vector_type(8)))  float    v8f;
typedef __attribute__((ext_vector_type(4)))  int      i4;

#define TM 128
#define TN 128
#define TK 32

#if __has_builtin(__builtin_amdgcn_global_load_async_to_lds_b128)
#define HAS_ASYNC_LDS 1
#else
#define HAS_ASYNC_LDS 0
#endif

__device__ __forceinline__ void async_cp16(const _Float16* g, _Float16* l)
{
#if HAS_ASYNC_LDS
    __builtin_amdgcn_global_load_async_to_lds_b128(
        (__attribute__((address_space(1))) i4*)g,
        (__attribute__((address_space(3))) i4*)l,
        0, 0);
#endif
}

__device__ __forceinline__ void wait_asynccnt_8()
{
#if __has_builtin(__builtin_amdgcn_s_wait_asynccnt)
    __builtin_amdgcn_s_wait_asynccnt(8);
#elif HAS_ASYNC_LDS
    asm volatile("s_wait_asynccnt 0x8" ::: "memory");
#endif
}

__device__ __forceinline__ void wait_asynccnt_0()
{
#if __has_builtin(__builtin_amdgcn_s_wait_asynccnt)
    __builtin_amdgcn_s_wait_asynccnt(0);
#elif HAS_ASYNC_LDS
    asm volatile("s_wait_asynccnt 0x0" ::: "memory");
#endif
}

// -------------------- LayerNorm + cast to f16 --------------------
// One block (256 threads) per row of 1024 floats.
__global__ __launch_bounds__(256)
void ln_cast_kernel(const float* __restrict__ x, const float* __restrict__ g,
                    const float* __restrict__ b, _Float16* __restrict__ y)
{
    const int row = blockIdx.x;
    const float* xr = x + (size_t)row * 1024;
    const int t = threadIdx.x;
    float4 v = ((const float4*)xr)[t];
    float s = v.x + v.y + v.z + v.w;
    float q = v.x * v.x + v.y * v.y + v.z * v.z + v.w * v.w;
    #pragma unroll
    for (int off = 16; off > 0; off >>= 1) {
        s += __shfl_xor(s, off, 32);
        q += __shfl_xor(q, off, 32);
    }
    __shared__ float ss[8], sq[8];
    const int wid = t >> 5, lane = t & 31;
    if (lane == 0) { ss[wid] = s; sq[wid] = q; }
    __syncthreads();
    if (t == 0) {
        float a = 0.f, c = 0.f;
        #pragma unroll
        for (int i = 0; i < 8; ++i) { a += ss[i]; c += sq[i]; }
        float m = a * (1.0f / 1024.0f);
        ss[0] = m;
        sq[0] = rsqrtf(c * (1.0f / 1024.0f) - m * m + 1e-5f);
    }
    __syncthreads();
    const float mean = ss[0], rstd = sq[0];
    float4 gg = ((const float4*)g)[t];
    float4 bb = ((const float4*)b)[t];
    h4 o;
    o[0] = (_Float16)((v.x - mean) * rstd * gg.x + bb.x);
    o[1] = (_Float16)((v.y - mean) * rstd * gg.y + bb.y);
    o[2] = (_Float16)((v.z - mean) * rstd * gg.z + bb.z);
    o[3] = (_Float16)((v.w - mean) * rstd * gg.w + bb.w);
    ((h4*)y)[(size_t)row * 256 + t] = o;
}

// -------------------- f32 -> f16 cast (weights) --------------------
__global__ __launch_bounds__(256)
void cast_f32_f16(const float* __restrict__ x, _Float16* __restrict__ y)
{
    const size_t i = (size_t)blockIdx.x * blockDim.x + threadIdx.x;
    float4 v = ((const float4*)x)[i];
    h4 o;
    o[0] = (_Float16)v.x; o[1] = (_Float16)v.y;
    o[2] = (_Float16)v.z; o[3] = (_Float16)v.w;
    ((h4*)y)[i] = o;
}

// -------------------- row softmax (2048 wide) -> f16 --------------------
__global__ __launch_bounds__(256)
void softmax_kernel(const float* __restrict__ S, _Float16* __restrict__ P)
{
    const int row = blockIdx.x;
    const float* sr = S + (size_t)row * 2048;
    const int t = threadIdx.x;
    float vals[8];
    float m = -1e30f;
    #pragma unroll
    for (int i = 0; i < 8; ++i) { vals[i] = sr[t + i * 256]; m = fmaxf(m, vals[i]); }
    #pragma unroll
    for (int off = 16; off > 0; off >>= 1) m = fmaxf(m, __shfl_xor(m, off, 32));
    __shared__ float red[8];
    __shared__ float bcast[2];
    const int wid = t >> 5, lane = t & 31;
    if (lane == 0) red[wid] = m;
    __syncthreads();
    if (t == 0) {
        float mm = red[0];
        #pragma unroll
        for (int i = 1; i < 8; ++i) mm = fmaxf(mm, red[i]);
        bcast[0] = mm;
    }
    __syncthreads();
    const float rowmax = bcast[0];
    __syncthreads();   // protect red[] reuse
    float s = 0.f;
    #pragma unroll
    for (int i = 0; i < 8; ++i) { vals[i] = __expf(vals[i] - rowmax); s += vals[i]; }
    #pragma unroll
    for (int off = 16; off > 0; off >>= 1) s += __shfl_xor(s, off, 32);
    if (lane == 0) red[wid] = s;
    __syncthreads();
    if (t == 0) {
        float a = 0.f;
        #pragma unroll
        for (int i = 0; i < 8; ++i) a += red[i];
        bcast[1] = 1.0f / a;
    }
    __syncthreads();
    const float inv = bcast[1];
    _Float16* pr = P + (size_t)row * 2048;
    #pragma unroll
    for (int i = 0; i < 8; ++i) pr[t + i * 256] = (_Float16)(vals[i] * inv);
}

// -------------------- WMMA inner compute step --------------------
// Fragment layouts per CDNA5 ISA 16-bit A/B VGPR tables.
__device__ __forceinline__ void mma_step(const _Float16* As, const _Float16* Bs,
                                         int wm, int wn, int lr, int kh, int kb,
                                         v8f (&acc)[4][4])
{
    v16h af[4], bf[4];
    #pragma unroll
    for (int f = 0; f < 4; ++f) {
        int ra = wm + f * 16 + lr;
        ((h8*)&af[f])[0] = *(const h8*)&As[ra * TK + kh];
        ((h8*)&af[f])[1] = *(const h8*)&As[ra * TK + kh + 16];
        int rb = wn + f * 16 + lr;
        ((h8*)&bf[f])[0] = *(const h8*)&Bs[rb * TK + kh];
        ((h8*)&bf[f])[1] = *(const h8*)&Bs[rb * TK + kh + 16];
    }
    #pragma unroll
    for (int mf = 0; mf < 4; ++mf)
        #pragma unroll
        for (int nf = 0; nf < 4; ++nf)
            acc[mf][nf] = __builtin_amdgcn_wmma_f32_16x16x32_f16(
                false, af[mf], false, bf[nf], (short)0, acc[mf][nf], false, false);
}

// -------------------- WMMA GEMM --------------------
// Per z-batch:  Y[M,N] = A[M,K] * B^T   where B is [N,K] row-major.
// BTRANS=true: B is [K,N] row-major instead (transposed while staging to LDS).
// MODE 0: store f16.  MODE 1: store relu(acc)*scale f32.  MODE 2: store acc+bias[n] f32.
template<int MODE, bool BTRANS>
__global__ __launch_bounds__(128)
void gemm_wmma_kernel(const _Float16* __restrict__ A, int lda, long long sA,
                      const _Float16* __restrict__ B, int ldb, long long sB,
                      void* __restrict__ Cv, int ldc, long long sC,
                      int M, int N, int K,
                      const float* __restrict__ bias, float scale)
{
    constexpr bool USE_ASYNC = (HAS_ASYNC_LDS != 0) && !BTRANS;
    constexpr int  NBUF      = USE_ASYNC ? 2 : 1;

    __shared__ __attribute__((aligned(16))) _Float16 As[NBUF][TM * TK];
    __shared__ __attribute__((aligned(16))) _Float16 Bs[NBUF][TN * TK];   // stored [n][k]

    const int bz = blockIdx.z;
    A += (size_t)bz * sA;
    B += (size_t)bz * sB;
    const int n0 = blockIdx.x * TN;
    const int m0 = blockIdx.y * TM;
    const int t = threadIdx.x;
    const int lane = t & 31;
    const int w = t >> 5;
    const int wm = (w >> 1) * 64;          // 2x2 wave grid, 64x64 per wave
    const int wn = (w & 1) * 64;
    const int lr = lane & 15;
    const int kh = (lane >> 4) * 8;        // A-fragment K split  {kh..kh+7, kh+16..kh+23}
    const int kb = (lane >> 4) * 16;       // unused: B fragment also loads {kh..kh+7, kh+16..kh+23}

    v8f acc[4][4];
    #pragma unroll
    for (int mf = 0; mf < 4; ++mf)
        #pragma unroll
        for (int nf = 0; nf < 4; ++nf)
            acc[mf][nf] = (v8f){};

    const int nk = K / TK;

    if constexpr (USE_ASYNC) {
        // -------- double-buffered, async-to-LDS staging (ASYNCcnt pipeline) --------
        auto issue_tile = [&](int kt, int buf) {
            const _Float16* Ag = A + (size_t)m0 * lda + kt * TK;
            #pragma unroll
            for (int i = 0; i < 4; ++i) {
                int ch = t + i * 128;
                int r = ch >> 2, c = (ch & 3) * 8;
                async_cp16(Ag + (size_t)r * lda + c, &As[buf][r * TK + c]);
            }
            const _Float16* Bg = B + (size_t)n0 * ldb + kt * TK;
            #pragma unroll
            for (int i = 0; i < 4; ++i) {
                int ch = t + i * 128;
                int r = ch >> 2, c = (ch & 3) * 8;
                async_cp16(Bg + (size_t)r * ldb + c, &Bs[buf][r * TK + c]);
            }
        };

        issue_tile(0, 0);
        for (int kt = 0; kt < nk; ++kt) {
            if (kt + 1 < nk) {
                issue_tile(kt + 1, (kt + 1) & 1);
                wait_asynccnt_8();      // retire tile kt's 8 copies; keep kt+1 in flight
            } else {
                wait_asynccnt_0();
            }
            __syncthreads();
            mma_step(As[kt & 1], Bs[kt & 1], wm, wn, lr, kh, kb, acc);
            __syncthreads();            // before buffer (kt+2)&1 gets overwritten
        }
    } else {
        // -------- synchronous staging (and transpose path for BTRANS) --------
        for (int kt = 0; kt < nk; ++kt) {
            const _Float16* Ag = A + (size_t)m0 * lda + kt * TK;
            #pragma unroll
            for (int i = 0; i < 4; ++i) {
                int ch = t + i * 128;
                int r = ch >> 2, c = (ch & 3) * 8;
                *(h8*)&As[0][r * TK + c] = *(const h8*)(Ag + (size_t)r * lda + c);
            }
            if (kt + 1 < nk)  // pull next A tile toward the caches (global_prefetch_b8)
                __builtin_prefetch(Ag + (size_t)(t >> 2) * lda + TK, 0, 3);

            if (!BTRANS) {
                const _Float16* Bg = B + (size_t)n0 * ldb + kt * TK;
                #pragma unroll
                for (int i = 0; i < 4; ++i) {
                    int ch = t + i * 128;
                    int r = ch >> 2, c = (ch & 3) * 8;
                    *(h8*)&Bs[0][r * TK + c] = *(const h8*)(Bg + (size_t)r * ldb + c);
                }
            } else {
                const _Float16* Bg = B + (size_t)(kt * TK) * ldb + n0;
                #pragma unroll
                for (int i = 0; i < 4; ++i) {
                    int ch = t + i * 128;            // 512 chunks of 8 halves
                    int r = ch >> 4;                 // k row 0..31
                    int cc = (ch & 15) * 8;          // n col
                    h8 vv = *(const h8*)(Bg + (size_t)r * ldb + cc);
                    #pragma unroll
                    for (int j = 0; j < 8; ++j) Bs[0][(cc + j) * TK + r] = vv[j];
                }
            }
            __syncthreads();
            mma_step(As[0], Bs[0], wm, wn, lr, kh, kb, acc);
            __syncthreads();
        }
    }

    // ---- epilogue: D layout = lane n%16, VGPR r -> m = r + 8*(lane/16) ----
    const int lm8 = (lane >> 4) * 8;
    #pragma unroll
    for (int mf = 0; mf < 4; ++mf) {
        #pragma unroll
        for (int nf = 0; nf < 4; ++nf) {
            const int n = n0 + wn + nf * 16 + lr;
            #pragma unroll
            for (int r2 = 0; r2 < 8; ++r2) {
                const int m = m0 + wm + mf * 16 + lm8 + r2;
                const float val = acc[mf][nf][r2];
                if (MODE == 0) {
                    ((_Float16*)Cv)[(size_t)bz * sC + (size_t)m * ldc + n] = (_Float16)val;
                } else if (MODE == 1) {
                    ((float*)Cv)[(size_t)bz * sC + (size_t)m * ldc + n] = fmaxf(val, 0.0f) * scale;
                } else {
                    ((float*)Cv)[(size_t)bz * sC + (size_t)m * ldc + n] = val + bias[n];
                }
            }
        }
    }
}

// -------------------- host orchestration --------------------
extern "C" void kernel_launch(void* const* d_in, const int* in_sizes, int n_in,
                              void* d_out, int out_size, void* d_ws, size_t ws_size,
                              hipStream_t stream)
{
    (void)in_sizes; (void)n_in; (void)out_size; (void)ws_size;
    const float* q  = (const float*)d_in[0];
    const float* k  = (const float*)d_in[1];
    const float* v  = (const float*)d_in[2];
    const float* gq = (const float*)d_in[3];
    const float* bq = (const float*)d_in[4];
    const float* gk = (const float*)d_in[5];
    const float* bk = (const float*)d_in[6];
    const float* gv = (const float*)d_in[7];
    const float* bv = (const float*)d_in[8];
    const float* Wq = (const float*)d_in[9];
    const float* Wk = (const float*)d_in[10];
    const float* Wv = (const float*)d_in[11];
    const float* Wp = (const float*)d_in[12];
    const float* bp = (const float*)d_in[13];

    const int Bb = 4, L = 2048, D = 1024;
    const size_t BL = (size_t)Bb * L;          // 8192

    _Float16* lnq = (_Float16*)d_ws;           // [BL, D] each
    _Float16* lnk = lnq + BL * D;
    _Float16* lnv = lnk + BL * D;
    _Float16* qn  = lnv + BL * D;
    _Float16* kn  = qn  + BL * D;
    _Float16* vn  = kn  + BL * D;
    _Float16* wq16 = vn + BL * D;              // [D, D] each
    _Float16* wk16 = wq16 + (size_t)D * D;
    _Float16* wv16 = wk16 + (size_t)D * D;
    _Float16* wp16 = wv16 + (size_t)D * D;
    _Float16* P = lnq;                         // reuse: [B, L, L] f16 = lnq+lnk region
    _Float16* O = lnv;                         // reuse: [BL, D] f16

    float* attn = (float*)d_out;               // [B, L, L]
    float* out  = attn + (size_t)Bb * L * L;   // [BL, D]

    // 1) LayerNorm + cast
    ln_cast_kernel<<<(int)BL, 256, 0, stream>>>(q, gq, bq, lnq);
    ln_cast_kernel<<<(int)BL, 256, 0, stream>>>(k, gk, bk, lnk);
    ln_cast_kernel<<<(int)BL, 256, 0, stream>>>(v, gv, bv, lnv);

    // 2) weight casts (D*D/4 threads each)
    const int cblocks = (D * D / 4) / 256;
    cast_f32_f16<<<cblocks, 256, 0, stream>>>(Wq, wq16);
    cast_f32_f16<<<cblocks, 256, 0, stream>>>(Wk, wk16);
    cast_f32_f16<<<cblocks, 256, 0, stream>>>(Wv, wv16);
    cast_f32_f16<<<cblocks, 256, 0, stream>>>(Wp, wp16);

    // 3) input projections: [BL,D] = LN(x) @ W^T
    dim3 gproj(D / TN, (int)(BL / TM), 1);
    gemm_wmma_kernel<0, false><<<gproj, 128, 0, stream>>>(lnq, D, 0, wq16, D, 0, qn, D, 0, (int)BL, D, D, nullptr, 1.0f);
    gemm_wmma_kernel<0, false><<<gproj, 128, 0, stream>>>(lnk, D, 0, wk16, D, 0, kn, D, 0, (int)BL, D, D, nullptr, 1.0f);
    gemm_wmma_kernel<0, false><<<gproj, 128, 0, stream>>>(lnv, D, 0, wv16, D, 0, vn, D, 0, (int)BL, D, D, nullptr, 1.0f);

    // 4) attention scores: attn[b] = relu(qn[b] @ kn[b]^T) * D^-0.5   (f32 -> d_out)
    dim3 gattn(L / TN, L / TM, Bb);
    gemm_wmma_kernel<1, false><<<gattn, 128, 0, stream>>>(
        qn, D, (long long)L * D, kn, D, (long long)L * D,
        attn, L, (long long)L * L, L, L, D, nullptr, 0.03125f /* 1/sqrt(1024) */);

    // 5) softmax over k axis -> f16 probabilities
    softmax_kernel<<<(int)BL, 256, 0, stream>>>(attn, P);

    // 6) O[b] = P[b] @ vn[b]   (B operand is [K,N] row-major -> transpose staging)
    dim3 gpv(D / TN, L / TM, Bb);
    gemm_wmma_kernel<0, true><<<gpv, 128, 0, stream>>>(
        P, L, (long long)L * L, vn, D, (long long)L * D,
        O, D, (long long)L * D, L, D, L, nullptr, 1.0f);

    // 7) output projection + bias: out = O @ Wp^T + bp  (f32 -> d_out)
    gemm_wmma_kernel<2, false><<<gproj, 128, 0, stream>>>(
        O, D, 0, wp16, D, 0, out, D, 0, (int)BL, D, D, bp, 1.0f);
}
